// RPN_45981919871046
// MI455X (gfx1250) — compile-verified, matrix operand order checked by reading, not run on hardware
//
#include <hip/hip_runtime.h>
#include <stdint.h>

typedef __attribute__((ext_vector_type(16))) _Float16 v16h;
typedef __attribute__((ext_vector_type(8)))  float    v8f;

#define C_IN    256
#define KDIM    2304          // 256 * 9
#define NTOT    270072        // anchors per image
#define NSORT   524288        // 2^19 >= NTOT
#define LOG2N   19
#define PREV_N  6000
#define POST_N  1000
#define NMS_THR 0.7f
#define TOT_PIX 180048        // sum over levels of B*H*W
#define SROW    272           // LDS row pitch in halves (256 + 16 pad)
#define SBUF    (64 * SROW)   // one staging buffer (halves)

__device__ __forceinline__ v16h pack16(uint4 a, uint4 b) {
  union { uint4 u[2]; v16h h; } x;
  x.u[0] = a; x.u[1] = b; return x.h;
}

// ---------------------------------------------------------------------------
// Weight transform: OIHW f32 -> wT[co][pos][ci] f16  (K = pos*256 + ci)
// ---------------------------------------------------------------------------
__global__ void conv_w_to_f16(const float* __restrict__ wsrc,
                              _Float16* __restrict__ wT) {
  int tid = blockIdx.x * blockDim.x + threadIdx.x;
  if (tid >= 256 * KDIM) return;
  int co  = tid / KDIM;
  int rem = tid - co * KDIM;
  int pos = rem >> 8;
  int ci  = rem & 255;
  int kh = pos / 3, kw = pos - kh * 3;
  wT[tid] = (_Float16)wsrc[(((size_t)co * 256 + ci) * 3 + kh) * 3 + kw];
}

__global__ void head_w_to_f16(const float* __restrict__ wc,
                              const float* __restrict__ wb,
                              _Float16* __restrict__ hc,
                              _Float16* __restrict__ hb) {
  int tid = blockIdx.x * blockDim.x + threadIdx.x;
  if (tid < 6 * 256)  hc[tid] = (_Float16)wc[tid];
  if (tid < 12 * 256) hb[tid] = (_Float16)wb[tid];
}

// zero the out-of-bounds tap page (512 bytes)
__global__ void zero_pad(float* __restrict__ zp) {
  int tid = threadIdx.x;
  if (tid < 128) zp[tid] = 0.f;
}

// NCHW f32 -> NHWC f16 (per level)
__global__ void to_nhwc_f16(const float* __restrict__ x,
                            _Float16* __restrict__ xT, int HW) {
  int tid = blockIdx.x * blockDim.x + threadIdx.x;
  if (tid >= 2 * HW * C_IN) return;
  int c = tid & 255;
  int rest = tid >> 8;
  int p = rest % HW;
  int b = rest / HW;
  xT[tid] = (_Float16)x[((size_t)b * C_IN + c) * HW + p];
}

// ---------------------------------------------------------------------------
// 3x3 conv (pad 1) + bias + ReLU, implicit GEMM via WMMA f32_16x16x32_f16.
// Block = 4 waves = 64 pixels x 64 out-channels (grid.y = channel group).
// Per tap: stage 64co x 256ci f16 weights (32KB) into a double-buffered LDS
// panel shared by all 4 waves; inner loop reads B via ds_load_b128 and A via
// global_load_b128 (register double buffer). Barriers pin the pipeline.
// xT: [b][p][ci] f16; wT: [co][pos*256+ci] f16; t: [b][p][co] f16.
// ---------------------------------------------------------------------------
__global__ __launch_bounds__(128)
void conv3x3_relu_wmma(const _Float16* __restrict__ xT,
                       const _Float16* __restrict__ wT,
                       const float* __restrict__ bias,
                       const _Float16* __restrict__ zpad,
                       _Float16* __restrict__ t,
                       int H, int W, int HW) {
  extern __shared__ _Float16 smB[];     // [2][SBUF]
  const int tidb = threadIdx.x;
  const int lane = tidb & 31;
  const int wave = tidb >> 5;           // 0..3 -> M sub-tile
  const int row  = lane & 15;
  const int hi   = lane >> 4;
  const int hk   = hi * 8;
  const int coBase = blockIdx.y * 64;   // channel group
  const int b      = blockIdx.z;
  const int tileM  = blockIdx.x * 4 + wave;
  const int m      = tileM * 16 + row;
  const bool mval  = (m < HW);
  const int h = m / W;
  const int w = m - h * W;
  const _Float16* xb = xT + (size_t)b * HW * C_IN;
  const _Float16* wg = wT + (size_t)coBase * KDIM;

  // Per-lane input row pointer for each of the 9 taps; OOB -> zero page.
  const _Float16* rowp[9];
#pragma unroll
  for (int pos = 0; pos < 9; ++pos) {
    const int dh = pos / 3 - 1, dw = pos % 3 - 1;
    const int hh = h + dh, ww = w + dw;
    const bool v = mval && (hh >= 0) && (hh < H) && (ww >= 0) && (ww < W);
    rowp[pos] = v ? (xb + (size_t)(hh * W + ww) * C_IN) : zpad;
  }
#pragma unroll
  for (int pos = 0; pos < 9; ++pos) __builtin_prefetch(rowp[pos], 0, 0);

  // cooperative stage of one tap's weight panel into LDS buffer `buf`
  auto stage = [&](int pos, int buf) {
    _Float16* dst = smB + (size_t)buf * SBUF;
#pragma unroll
    for (int k = 0; k < 16; ++k) {
      const int chunk = tidb * 16 + k;        // 2048 x 16B chunks
      const int co  = chunk >> 5;             // 32 chunks per row
      const int cio = (chunk & 31) * 8;
      uint4 v = *(const uint4*)(wg + (size_t)co * KDIM + pos * 256 + cio);
      *(uint4*)(dst + co * SROW + cio) = v;
    }
  };

  v8f acc[4] = {};

  stage(0, 0);
  __syncthreads();

#pragma unroll
  for (int pos = 0; pos < 9; ++pos) {
    const int buf = pos & 1;
    if (pos + 1 < 9) stage(pos + 1, buf ^ 1);   // fill spare buffer
    const _Float16* rp = rowp[pos];
    const _Float16* sb = smB + (size_t)buf * SBUF;

    // A register double buffer across the 8 ci-blocks
    uint4 a0 = *(const uint4*)(rp + hk);
    uint4 a1 = *(const uint4*)(rp + hk + 16);
#pragma unroll
    for (int cb = 0; cb < 8; ++cb) {
      uint4 na0 = a0, na1 = a1;
      if (cb + 1 < 8) {
        na0 = *(const uint4*)(rp + (cb + 1) * 32 + hk);
        na1 = *(const uint4*)(rp + (cb + 1) * 32 + hk + 16);
      }
      const v16h A = pack16(a0, a1);
#pragma unroll
      for (int n = 0; n < 4; ++n) {
        const _Float16* bp = sb + (n * 16 + row) * SROW + cb * 32 + hk;
        uint4 b0 = *(const uint4*)(bp);
        uint4 b1 = *(const uint4*)(bp + 16);
        acc[n] = __builtin_amdgcn_wmma_f32_16x16x32_f16(
            false, A, false, pack16(b0, b1), (short)0, acc[n], false, false);
      }
      a0 = na0; a1 = na1;
    }
    __syncthreads();   // staged (pos+1) panel complete; readers of `buf` done
  }

  // C/D layout: VGPR r holds M = hi*8 + r, column N = lane&15
#pragma unroll
  for (int n = 0; n < 4; ++n) {
    const int co = coBase + n * 16 + row;
    const float bv = bias[co];
#pragma unroll
    for (int r = 0; r < 8; ++r) {
      const int m2 = tileM * 16 + hi * 8 + r;
      if (m2 < HW) {
        float vv = acc[n][r] + bv;
        vv = vv > 0.f ? vv : 0.f;
        t[((size_t)b * HW + m2) * C_IN + co] = (_Float16)vv;
      }
    }
  }
}

// ---------------------------------------------------------------------------
// 1x1 heads + softmax + anchor gen + decode + clip + min-size filter.
// One thread per (b, pixel).
// ---------------------------------------------------------------------------
__global__ void rpn_heads(const _Float16* __restrict__ t,
                          const _Float16* __restrict__ wcls,
                          const _Float16* __restrict__ wbbox,
                          const float* __restrict__ bcls,
                          const float* __restrict__ bbbox,
                          const float* __restrict__ im_info,
                          float* __restrict__ probs,
                          float* __restrict__ boxes,
                          int H, int W, int HW, int off_stride, int anchorBase) {
  int tid = blockIdx.x * blockDim.x + threadIdx.x;
  if (tid >= 2 * HW) return;
  int b = tid / HW;
  int p = tid - b * HW;
  int h = p / W, w = p - h * W;
  const _Float16* tp = t + ((size_t)b * HW + p) * C_IN;

  float ac[6]  = {0.f, 0.f, 0.f, 0.f, 0.f, 0.f};
  float ab[12] = {0.f, 0.f, 0.f, 0.f, 0.f, 0.f, 0.f, 0.f, 0.f, 0.f, 0.f, 0.f};
  for (int c = 0; c < C_IN; ++c) {
    float tv = (float)tp[c];
#pragma unroll
    for (int o = 0; o < 6; ++o)  ac[o] += tv * (float)wcls[o * C_IN + c];
#pragma unroll
    for (int o = 0; o < 12; ++o) ab[o] += tv * (float)wbbox[o * C_IN + c];
  }

  const float imh = im_info[b * 6 + 0];
  const float imw = im_info[b * 6 + 1];
  const float imsc = im_info[b * 6 + 2];
  const float stride = 4.0f * (float)off_stride;
  const float sx = (float)w * stride, sy = (float)h * stride;
  const float scale = 2.0f * (float)off_stride;
  const float ratios[3] = {1.f, 2.f, 3.f};

  for (int a = 0; a < 3; ++a) {
    float ws = roundf(sqrtf(256.f / ratios[a]));
    float hs = roundf(ws * ratios[a]);
    float wss = ws * scale, hss = hs * scale;
    const float ctr = 7.5f;
    float ax1 = ctr - 0.5f * (wss - 1.f) + sx;
    float ay1 = ctr - 0.5f * (hss - 1.f) + sy;
    float ax2 = ctr + 0.5f * (wss - 1.f) + sx;
    float ay2 = ctr + 0.5f * (hss - 1.f) + sy;

    float l0 = ac[2 * a] + bcls[2 * a];
    float l1 = ac[2 * a + 1] + bcls[2 * a + 1];
    float prob = 1.f / (1.f + expf(l0 - l1));

    float dx = ab[4 * a + 0] + bbbox[4 * a + 0];
    float dy = ab[4 * a + 1] + bbbox[4 * a + 1];
    float dwv = ab[4 * a + 2] + bbbox[4 * a + 2];
    float dhv = ab[4 * a + 3] + bbbox[4 * a + 3];

    float aw = ax2 - ax1 + 1.f, ah = ay2 - ay1 + 1.f;
    float axc = ax1 + 0.5f * aw, ayc = ay1 + 0.5f * ah;
    float px = axc + dx * aw, py = ayc + dy * ah;
    float pw = aw * expf(dwv), ph = ah * expf(dhv);
    float x1 = px - 0.5f * pw, y1 = py - 0.5f * ph;
    float x2 = px + 0.5f * pw, y2 = py + 0.5f * ph;
    x1 = fminf(fmaxf(x1, 0.f), imw - 1.f);
    y1 = fminf(fmaxf(y1, 0.f), imh - 1.f);
    x2 = fminf(fmaxf(x2, 0.f), imw - 1.f);
    y2 = fminf(fmaxf(y2, 0.f), imh - 1.f);
    float thr = 2.0f * imsc;
    bool ok = ((x2 - x1 + 1.f) >= thr) && ((y2 - y1 + 1.f) >= thr);
    float sc = ok ? prob : -1.f;

    size_t gi = (size_t)b * NTOT + (size_t)anchorBase + (size_t)p * 3 + a;
    probs[gi] = sc;
    float* bx = boxes + gi * 4;
    bx[0] = x1; bx[1] = y1; bx[2] = x2; bx[3] = y2;
  }
}

// ---------------------------------------------------------------------------
// Sort keys: descending score, stable by ascending index
// ---------------------------------------------------------------------------
__global__ void init_keys(const float* __restrict__ probs,
                          unsigned long long* __restrict__ keys) {
  int tid = blockIdx.x * blockDim.x + threadIdx.x;
  if (tid >= 2 * NSORT) return;
  int img = tid >> LOG2N;
  int i = tid & (NSORT - 1);
  unsigned long long key = 0ull;
  if (i < NTOT) {
    float s = probs[(size_t)img * NTOT + i];
    unsigned ub = __float_as_uint(s);
    ub = (ub & 0x80000000u) ? ~ub : (ub | 0x80000000u);
    key = ((unsigned long long)ub << 32) |
          (unsigned long long)(0xFFFFFFFFu - (unsigned)i);
  }
  keys[tid] = key;
}

__global__ void bitonic_pass(unsigned long long* __restrict__ keys,
                             int j, int k) {
  int tid = blockIdx.x * blockDim.x + threadIdx.x;
  if (tid >= 2 * NSORT) return;
  int img = tid >> LOG2N;
  int i = tid & (NSORT - 1);
  int ixj = i ^ j;
  if (ixj > i) {
    unsigned long long* kk = keys + ((size_t)img << LOG2N);
    unsigned long long a = kk[i], b = kk[ixj];
    bool up = ((i & k) == 0);   // descending sort overall
    if (up ? (a < b) : (a > b)) { kk[i] = b; kk[ixj] = a; }
  }
}

__global__ void gather_topk(const unsigned long long* __restrict__ keys,
                            const float* __restrict__ probs,
                            const float* __restrict__ boxes,
                            float* __restrict__ topb,
                            float* __restrict__ areas,
                            int* __restrict__ alive) {
  int tid = blockIdx.x * blockDim.x + threadIdx.x;
  if (tid >= 2 * PREV_N) return;
  int img = tid / PREV_N;
  unsigned long long key = keys[((size_t)img << LOG2N) + (tid - img * PREV_N)];
  unsigned idx = 0xFFFFFFFFu - (unsigned)(key & 0xFFFFFFFFull);
  size_t gi = (size_t)img * NTOT + idx;
  float sc = probs[gi];
  float x1 = boxes[gi * 4 + 0], y1 = boxes[gi * 4 + 1];
  float x2 = boxes[gi * 4 + 2], y2 = boxes[gi * 4 + 3];
  topb[(size_t)tid * 4 + 0] = x1;
  topb[(size_t)tid * 4 + 1] = y1;
  topb[(size_t)tid * 4 + 2] = x2;
  topb[(size_t)tid * 4 + 3] = y2;
  areas[tid] = (x2 - x1 + 1.f) * (y2 - y1 + 1.f);
  alive[tid] = (sc > -0.5f) ? 1 : 0;
}

// ---------------------------------------------------------------------------
// Greedy NMS, one block per image. Scores sorted -> first-alive is monotone.
// ---------------------------------------------------------------------------
__global__ __launch_bounds__(256)
void nms_kernel(const float* __restrict__ topb,
                const float* __restrict__ areas,
                const int* __restrict__ aliveInit,
                float* __restrict__ out) {
  const int img = blockIdx.x;
  const float* tb = topb + (size_t)img * PREV_N * 4;
  const float* ar = areas + (size_t)img * PREV_N;
  const int* ai = aliveInit + (size_t)img * PREV_N;

  __shared__ unsigned mask[(PREV_N + 31) / 32];
  __shared__ int s_i;
  __shared__ float s_box[4];
  __shared__ float s_area;
  __shared__ int s_cursor;

  const int tid = threadIdx.x;
  for (int wi = tid; wi < (PREV_N + 31) / 32; wi += blockDim.x) {
    unsigned v = 0u;
    for (int bit = 0; bit < 32; ++bit) {
      int s = wi * 32 + bit;
      if (s < PREV_N && ai[s]) v |= (1u << bit);
    }
    mask[wi] = v;
  }
  if (tid == 0) s_cursor = 0;
  __syncthreads();

  for (int it = 0; it < POST_N; ++it) {
    if (tid == 0) {
      int c = s_cursor;
      while (c < PREV_N && !((mask[c >> 5] >> (c & 31)) & 1u)) ++c;
      s_cursor = c;
      if (c < PREV_N) {
        s_i = c;
        s_box[0] = tb[c * 4 + 0];
        s_box[1] = tb[c * 4 + 1];
        s_box[2] = tb[c * 4 + 2];
        s_box[3] = tb[c * 4 + 3];
        s_area = ar[c];
        mask[c >> 5] &= ~(1u << (c & 31));
      } else {
        s_i = -1;
      }
    }
    __syncthreads();

    float* orow = out + ((size_t)img * POST_N + it) * 5;
    if (s_i < 0) {
      if (tid < 5) orow[tid] = 0.f;
    } else {
      if (tid == 0) orow[0] = (float)img;
      if (tid >= 1 && tid <= 4) orow[tid] = s_box[tid - 1];
      const float bx1 = s_box[0], by1 = s_box[1];
      const float bx2 = s_box[2], by2 = s_box[3], ba = s_area;
      for (int j = tid; j < PREV_N; j += (int)blockDim.x) {
        if ((mask[j >> 5] >> (j & 31)) & 1u) {
          float ix1 = fmaxf(bx1, tb[j * 4 + 0]);
          float iy1 = fmaxf(by1, tb[j * 4 + 1]);
          float ix2 = fminf(bx2, tb[j * 4 + 2]);
          float iy2 = fminf(by2, tb[j * 4 + 3]);
          float iw = fmaxf(ix2 - ix1 + 1.f, 0.f);
          float ih = fmaxf(iy2 - iy1 + 1.f, 0.f);
          float inter = iw * ih;
          float iou = inter / (ba + ar[j] - inter);
          if (iou > NMS_THR) atomicAnd(&mask[j >> 5], ~(1u << (j & 31)));
        }
      }
    }
    __syncthreads();
  }
}

// ---------------------------------------------------------------------------
extern "C" void kernel_launch(void* const* d_in, const int* in_sizes, int n_in,
                              void* d_out, int out_size, void* d_ws, size_t ws_size,
                              hipStream_t stream) {
  (void)in_sizes; (void)n_in; (void)out_size; (void)ws_size;

  static const int Hs[5]  = {12, 24, 48, 96, 192};
  static const int Ws5[5] = {22, 44, 88, 176, 352};
  static const int HWs[5] = {264, 1056, 4224, 16896, 67584};
  static const int pixBase[5] = {0, 528, 2640, 11088, 44880};   // B*HW cumulative
  static const int aBase[5]   = {0, 792, 3960, 16632, 67320};   // anchors cumulative
  static const int offS[5]    = {16, 8, 4, 2, 1};

  const float* feats[5] = {(const float*)d_in[0], (const float*)d_in[1],
                           (const float*)d_in[2], (const float*)d_in[3],
                           (const float*)d_in[4]};
  const float* rpn_w  = (const float*)d_in[5];
  const float* rpn_b  = (const float*)d_in[6];
  const float* cls_w  = (const float*)d_in[7];
  const float* cls_b  = (const float*)d_in[8];
  const float* bbox_w = (const float*)d_in[9];
  const float* bbox_b = (const float*)d_in[10];
  const float* iminfo = (const float*)d_in[11];
  float* out = (float*)d_out;

  // workspace layout (256B aligned)
  uint8_t* ws = (uint8_t*)d_ws;
  size_t o = 0;
  auto alloc = [&](size_t bytes) -> uint8_t* {
    uint8_t* p = ws + o;
    o = (o + bytes + 255) & ~(size_t)255;
    return p;
  };
  _Float16* wT    = (_Float16*)alloc((size_t)256 * KDIM * 2);
  _Float16* hc    = (_Float16*)alloc((size_t)6 * 256 * 2);
  _Float16* hb    = (_Float16*)alloc((size_t)12 * 256 * 2);
  float*    zpad  = (float*)alloc(512);
  _Float16* tbuf  = (_Float16*)alloc((size_t)TOT_PIX * C_IN * 2);
  float*    probs = (float*)alloc((size_t)2 * NTOT * 4);
  float*    boxes = (float*)alloc((size_t)2 * NTOT * 4 * 4);
  float*    topb  = (float*)alloc((size_t)2 * PREV_N * 4 * 4);
  float*    areas = (float*)alloc((size_t)2 * PREV_N * 4);
  int*      alive = (int*)alloc((size_t)2 * PREV_N * 4);
  uint8_t*  bigRegion = alloc((size_t)TOT_PIX * C_IN * 2);   // xT, later aliased by keys
  _Float16* xT = (_Float16*)bigRegion;
  unsigned long long* keys = (unsigned long long*)bigRegion; // 8MB << 92MB, xT dead by then

  // 1) weight transforms + zero page
  {
    int tot = 256 * KDIM;
    conv_w_to_f16<<<(tot + 255) / 256, 256, 0, stream>>>(rpn_w, wT);
    head_w_to_f16<<<(12 * 256 + 255) / 256, 256, 0, stream>>>(cls_w, bbox_w, hc, hb);
    zero_pad<<<1, 128, 0, stream>>>(zpad);
  }

  // 2) inputs -> NHWC f16
  for (int l = 0; l < 5; ++l) {
    int tot = 2 * HWs[l] * C_IN;
    to_nhwc_f16<<<(tot + 255) / 256, 256, 0, stream>>>(
        feats[l], xT + (size_t)pixBase[l] * C_IN, HWs[l]);
  }

  // 3) 3x3 conv + ReLU (WMMA implicit GEMM, LDS-staged weights)
  {
    const size_t ldsBytes = (size_t)2 * SBUF * sizeof(_Float16);  // ~68KB
    for (int l = 0; l < 5; ++l) {
      dim3 grid((HWs[l] + 63) / 64, 4, 2);
      conv3x3_relu_wmma<<<grid, 128, ldsBytes, stream>>>(
          xT + (size_t)pixBase[l] * C_IN, wT, rpn_b, (const _Float16*)zpad,
          tbuf + (size_t)pixBase[l] * C_IN, Hs[l], Ws5[l], HWs[l]);
    }
  }

  // 4) heads + softmax + decode + clip + filter
  for (int l = 0; l < 5; ++l) {
    int tot = 2 * HWs[l];
    rpn_heads<<<(tot + 255) / 256, 256, 0, stream>>>(
        tbuf + (size_t)pixBase[l] * C_IN, hc, hb, cls_b, bbox_b, iminfo,
        probs, boxes, Hs[l], Ws5[l], HWs[l], offS[l], aBase[l]);
  }

  // 5) bitonic sort (descending), both images in one grid
  {
    int tot = 2 * NSORT;
    init_keys<<<(tot + 255) / 256, 256, 0, stream>>>(probs, keys);
    for (int k = 2; k <= NSORT; k <<= 1)
      for (int j = k >> 1; j > 0; j >>= 1)
        bitonic_pass<<<(tot + 255) / 256, 256, 0, stream>>>(keys, j, k);
  }

  // 6) gather top-6000 and run NMS
  {
    int tot = 2 * PREV_N;
    gather_topk<<<(tot + 255) / 256, 256, 0, stream>>>(keys, probs, boxes,
                                                       topb, areas, alive);
    nms_kernel<<<2, 256, 0, stream>>>(topb, areas, alive, out);
  }
}
